// SimpleCrossAttention_3195455668699
// MI455X (gfx1250) — matrix operand order repaired and checked by reading, not compiled, not run on hardware
//
#include <hip/hip_runtime.h>

// ---------------------------------------------------------------------------
// SimpleCrossAttention for MI455X (gfx1250, wave32, WMMA)
// B=2, DIM=512, HIDDEN=512, HEADS=8, HEAD_DIM=64, S=H*W=4096
// bf16 WMMA everywhere; flash attention (no [S,S] materialization);
// weights pre-transposed so GEMM LDS path is pure b128; softmax row
// reductions via v_permlane16_b32; global->LDS copies via async-to-LDS
// (ASYNCcnt) where available; uniform-base + 32-bit offset addressing.
// ---------------------------------------------------------------------------

typedef __attribute__((ext_vector_type(16))) __bf16 v16bf;
typedef __attribute__((ext_vector_type(8)))  float  v8f;
typedef __attribute__((ext_vector_type(4)))  int    v4i;

__device__ inline v8f zero8() {
  v8f z;
  #pragma unroll
  for (int i = 0; i < 8; ++i) z[i] = 0.0f;
  return z;
}

__device__ inline v8f wmma_bf16(v16bf a, v16bf b, v8f c) {
  // D = A(16x32) * B(32x16) + C(16x16), f32 accumulate
  return __builtin_amdgcn_wmma_f32_16x16x32_bf16(
      false, a, false, b, (short)0, c, false, false);
}

// ---- async global->LDS copy (GLOBAL_LOAD_ASYNC_TO_LDS_B128) ---------------
#if defined(__has_builtin)
#if __has_builtin(__builtin_amdgcn_global_load_async_to_lds_b128) && \
    __has_builtin(__builtin_amdgcn_s_wait_asynccnt)
#define HAVE_ASYNC 1
#endif
#if __has_builtin(__builtin_amdgcn_permlane16)
#define USE_PERMLANE16 1
#endif
#endif

typedef __attribute__((address_space(1))) v4i gv4i;  // global int4
typedef __attribute__((address_space(3))) v4i lv4i;  // LDS int4

__device__ inline void cp16(const __bf16* g, __bf16* l) {
#ifdef HAVE_ASYNC
  __builtin_amdgcn_global_load_async_to_lds_b128(
      (gv4i*)g, (lv4i*)l, 0, 0);
#else
  *(uint4*)l = *(const uint4*)g;
#endif
}
__device__ inline void async_wait() {
#ifdef HAVE_ASYNC
  __builtin_amdgcn_s_wait_asynccnt(0);
#endif
}

// ---- row (16-lane) all-reduce via v_permlane16_b32 -------------------------
__device__ inline float row_shuffle_xor(float v, int m) {
#ifdef USE_PERMLANE16
  unsigned s1, s2;  // lane i gathers from lane (i^m) within its 16-lane row
  switch (m) {
    case 1: s1 = 0x67452301u; s2 = 0xEFCDAB89u; break;
    case 2: s1 = 0x54761032u; s2 = 0xDCFE98BAu; break;
    case 4: s1 = 0x32107654u; s2 = 0xBA98FEDCu; break;
    default: s1 = 0xFEDCBA98u; s2 = 0x76543210u; break; // m == 8
  }
  unsigned u = __builtin_bit_cast(unsigned, v);
  unsigned r = __builtin_amdgcn_permlane16(u, u, s1, s2, false, false);
  return __builtin_bit_cast(float, r);
#else
  return __shfl_xor(v, m, 32);
#endif
}

__device__ inline float row_max16(float v) {
  v = fmaxf(v, row_shuffle_xor(v, 1));
  v = fmaxf(v, row_shuffle_xor(v, 2));
  v = fmaxf(v, row_shuffle_xor(v, 4));
  v = fmaxf(v, row_shuffle_xor(v, 8));
  return v;
}
__device__ inline float row_sum16(float v) {
  v += row_shuffle_xor(v, 1);
  v += row_shuffle_xor(v, 2);
  v += row_shuffle_xor(v, 4);
  v += row_shuffle_xor(v, 8);
  return v;
}

union FragU { v16bf v; uint4 u[2]; };

// A-matrix 16x32 bf16 fragment from row-major LDS tile.
// ISA 7.12.2: lanes 0-15 -> K 0..7 & 16..23, lanes 16-31 -> K 8..15 & 24..31.
__device__ inline v16bf lds_load_a(const __bf16* base, int stride, int lane) {
  const int l16 = lane & 15, hf = lane >> 4;
  const __bf16* p = base + l16 * stride + hf * 8;
  FragU f;
  f.u[0] = *(const uint4*)(p);
  f.u[1] = *(const uint4*)(p + 16);
  return f.v;
}

// B-matrix 32x16 bf16 fragment from an LDS tile stored as [N][K].
__device__ inline v16bf lds_load_bt(const __bf16* base, int stride, int lane) {
  const int l16 = lane & 15, hf = lane >> 4;
  const __bf16* p = base + l16 * stride + hf * 8;
  FragU f;
  f.u[0] = *(const uint4*)(p);
  f.u[1] = *(const uint4*)(p + 16);
  return f.v;
}

// ---------------------------------------------------------------------------
// Conversion kernels
// ---------------------------------------------------------------------------

// src: [B, C, S] f32  ->  dst: [B, S, C] bf16   (C=512, S=4096)
__global__ __launch_bounds__(256) void transpose_cast_kernel(
    const float* __restrict__ src, __bf16* __restrict__ dst) {
  const long i = (long)blockIdx.x * blockDim.x + threadIdx.x; // c fastest
  const int c = (int)(i & 511);
  const long t = i >> 9;
  const int s = (int)(t & 4095);
  const int b = (int)(t >> 12);
  dst[i] = (__bf16)src[((long)b * 512 + c) * 4096 + s];
}

// weight src: [K=512, N=512] f32 -> dst: [N, K] bf16 (pre-transposed for GEMM B path)
__global__ __launch_bounds__(256) void weight_cast_transpose_kernel(
    const float* __restrict__ src, __bf16* __restrict__ dst) {
  const int i = blockIdx.x * blockDim.x + threadIdx.x; // k fastest
  const int k = i & 511, n = i >> 9;
  dst[i] = (__bf16)src[k * 512 + n];
}

// ---------------------------------------------------------------------------
// GEMM:  C[M,N] = act(A[M,K] @ W[K,N] + bias), W provided TRANSPOSED as Bt[N,K].
// Block tile 128x128, 256 threads = 8 waves (2x4); wave computes 64x32; K-step 32.
// ACT: 0=none, 1=exact GELU.  OUTF32T: 0 -> bf16 [M,N]; 1 -> f32 [B,N,S] (coalesced
// via LDS staging).
// ---------------------------------------------------------------------------
template <int ACT, int OUTF32T>
__global__ __launch_bounds__(256) void gemm_bf16_kernel(
    const __bf16* __restrict__ A, const __bf16* __restrict__ Bt,
    const float* __restrict__ bias, void* __restrict__ Cout,
    int M, int N, int K) {
  __shared__ __align__(16) __bf16 sA[128 * 40];   // [m][k], stride 40 (pad 8)
  __shared__ __align__(16) __bf16 sB[128 * 40];   // [n][k], stride 40

  const int tid = threadIdx.x;
  const int lane = tid & 31, wid = tid >> 5;
  const int wm = wid >> 2, wn = wid & 3;          // wave grid 2(M) x 4(N)
  const int m0 = blockIdx.y * 128, n0 = blockIdx.x * 128;

  // uniform bases + 32-bit per-thread offsets (saddr + voffset addressing)
  const __bf16* Abase = A + (size_t)m0 * K;
  const __bf16* Bbase = Bt + (size_t)n0 * K;
  const int lrow = tid >> 2, lch8 = (tid & 3) * 8;

  v8f acc[4][2];
  #pragma unroll
  for (int i = 0; i < 4; ++i)
    #pragma unroll
    for (int j = 0; j < 2; ++j) acc[i][j] = zero8();

  for (int kk = 0; kk < K; kk += 32) {
    #pragma unroll
    for (int it = 0; it < 2; ++it) {
      const int row = lrow + it * 64;
      const int goff = row * K + kk + lch8;       // 32-bit offset
      const int loff = row * 40 + lch8;
      cp16(Abase + goff, &sA[loff]);
      cp16(Bbase + goff, &sB[loff]);
    }
    if (kk + 32 < K) {  // prefetch next K-step tiles (global_prefetch_b8)
      __builtin_prefetch(Abase + lrow * K + kk + 32, 0, 1);
      __builtin_prefetch(Bbase + lrow * K + kk + 32, 0, 1);
    }
    async_wait();
    __syncthreads();

    v16bf bf0 = lds_load_bt(&sB[(wn * 32 + 0) * 40], 40, lane);
    v16bf bf1 = lds_load_bt(&sB[(wn * 32 + 16) * 40], 40, lane);
    #pragma unroll
    for (int mt = 0; mt < 4; ++mt) {
      v16bf af = lds_load_a(&sA[(wm * 64 + mt * 16) * 40], 40, lane);
      acc[mt][0] = wmma_bf16(af, bf0, acc[mt][0]);
      acc[mt][1] = wmma_bf16(af, bf1, acc[mt][1]);
    }
    __syncthreads();
  }

  // epilogue: C/D layout reg r -> row (r + hf*8), col l16 within 16x16 tile
  const int l16 = lane & 15, hf = lane >> 4;
  if constexpr (OUTF32T) {
    // stage f32 block in LDS as [n][m], then coalesced writes along s
    __shared__ __align__(16) float sOut[128 * 132];
    #pragma unroll
    for (int mt = 0; mt < 4; ++mt)
      #pragma unroll
      for (int nt = 0; nt < 2; ++nt) {
        const int n_in = wn * 32 + nt * 16 + l16;
        const float bv = bias ? bias[n0 + n_in] : 0.0f;
        #pragma unroll
        for (int r = 0; r < 8; ++r) {
          const int m_in = wm * 64 + mt * 16 + hf * 8 + r;
          float v = acc[mt][nt][r] + bv;
          if (ACT == 1) v = 0.5f * v * (1.0f + erff(v * 0.70710678118654752f));
          sOut[n_in * 132 + m_in] = v;
        }
      }
    __syncthreads();
    float* out = (float*)Cout;
    const int bb = m0 >> 12, s0 = m0 & 4095;      // S = 4096
    #pragma unroll
    for (int i = tid; i < 128 * 32; i += 256) {   // 128 c-rows x 32 float4 chunks
      const int n = i >> 5, ch = i & 31;
      *(float4*)&out[((size_t)bb * N + n0 + n) * 4096 + s0 + ch * 4] =
          *(float4*)&sOut[n * 132 + ch * 4];
    }
  } else {
    __bf16* outb = (__bf16*)Cout;
    #pragma unroll
    for (int mt = 0; mt < 4; ++mt)
      #pragma unroll
      for (int nt = 0; nt < 2; ++nt) {
        const int n_g = n0 + wn * 32 + nt * 16 + l16;
        const float bv = bias ? bias[n_g] : 0.0f;
        #pragma unroll
        for (int r = 0; r < 8; ++r) {
          const int m_g = m0 + wm * 64 + mt * 16 + hf * 8 + r;
          float v = acc[mt][nt][r] + bv;
          if (ACT == 1) v = 0.5f * v * (1.0f + erff(v * 0.70710678118654752f));
          outb[(size_t)m_g * N + n_g] = (__bf16)v;
        }
      }
  }
}

// ---------------------------------------------------------------------------
// Flash attention: grid (B*HEADS, S/64), 128 threads = 4 waves.
// Each wave owns 16 q-rows; streams 64-wide K/V chunks; online softmax.
// Q/K/V/O layout: [B*S, 512] bf16, head h at columns h*64..h*64+63.
// ---------------------------------------------------------------------------
__global__ __launch_bounds__(128) void attn_kernel(
    const __bf16* __restrict__ Qb, const __bf16* __restrict__ Kb,
    const __bf16* __restrict__ Vb, __bf16* __restrict__ Ob) {
  __shared__ __align__(16) __bf16 sQ[64 * 72];
  __shared__ __align__(16) __bf16 sK[64 * 72];   // [k][d]  (== B^T for QK^T)
  __shared__ __align__(16) __bf16 sVt[64 * 72];  // [d][k]  (transposed V)
  __shared__ __align__(16) __bf16 sP[64 * 72];   // [q][k]  probs relayout

  const int tid = threadIdx.x, lane = tid & 31, w = tid >> 5;
  const int l16 = lane & 15, hf = lane >> 4;
  const int b = blockIdx.x >> 3, h = blockIdx.x & 7;
  const int q0 = blockIdx.y * 64;
  const int colbase = h * 64;

  // uniform bases (b, head folded in); all per-thread offsets are 32-bit
  const size_t bhead = ((size_t)b << 21) + colbase;   // b*4096*512 + h*64
  const __bf16* Qbase = Qb + bhead;
  const __bf16* Kbase = Kb + bhead;
  const __bf16* Vbase = Vb + bhead;
  __bf16* Obase = (__bf16*)Ob + bhead;
  const int trow = tid >> 3;                          // 0..15
  const int tch8 = (tid & 7) * 8;                     // 0..56

  // load Q tile 64x64 (async copy)
  #pragma unroll
  for (int it = 0; it < 4; ++it) {
    const int row = trow + it * 16;
    cp16(Qbase + (q0 + row) * 512 + tch8, &sQ[row * 72 + tch8]);
  }
  async_wait();
  __syncthreads();

  const v16bf aq0 = lds_load_a(&sQ[(w * 16) * 72 + 0], 72, lane);
  const v16bf aq1 = lds_load_a(&sQ[(w * 16) * 72 + 32], 72, lane);

  float mrow[8], lrow[8];
  v8f acco[4];
  #pragma unroll
  for (int r = 0; r < 8; ++r) { mrow[r] = -1e30f; lrow[r] = 0.0f; }
  #pragma unroll
  for (int dt = 0; dt < 4; ++dt) acco[dt] = zero8();

  for (int k0 = 0; k0 < 4096; k0 += 64) {
    // load K chunk [64 k][64 d] (async copy)
    #pragma unroll
    for (int it = 0; it < 4; ++it) {
      const int row = trow + it * 16;
      cp16(Kbase + (k0 + row) * 512 + tch8, &sK[row * 72 + tch8]);
    }
    // load V chunk transposed -> sVt[d][k] (VGPR path: transpose on the way in)
    #pragma unroll
    for (int it = 0; it < 4; ++it) {
      const int row = trow + it * 16;
      union { uint4 u; __bf16 e[8]; } t;
      t.u = *(const uint4*)&Vbase[(k0 + row) * 512 + tch8];
      #pragma unroll
      for (int j = 0; j < 8; ++j) sVt[(tch8 + j) * 72 + row] = t.e[j];
    }
    if (k0 + 64 < 4096) {  // prefetch next chunk rows (one line per row)
      const int noff = (k0 + 64 + (tid >> 1)) * 512;
      __builtin_prefetch(Kbase + noff, 0, 1);
      __builtin_prefetch(Vbase + noff, 0, 1);
    }
    async_wait();
    __syncthreads();

    // S = (Q @ K^T) * SCALE : 4 tiles of 16x16, K-dim = d = 64 (2 WMMAs each)
    v8f sacc[4];
    #pragma unroll
    for (int nt = 0; nt < 4; ++nt) {
      v16bf bk0 = lds_load_bt(&sK[(nt * 16) * 72 + 0], 72, lane);
      v16bf bk1 = lds_load_bt(&sK[(nt * 16) * 72 + 32], 72, lane);
      v8f s = wmma_bf16(aq0, bk0, zero8());
      s = wmma_bf16(aq1, bk1, s);
      #pragma unroll
      for (int r = 0; r < 8; ++r) s[r] *= 0.125f;  // SCALE = 64^-0.5
      sacc[nt] = s;
    }

    // online softmax: per-row (16-lane) reductions via permlane16
    float mnew[8], alpha[8], psum[8];
    #pragma unroll
    for (int r = 0; r < 8; ++r) {
      float cm = fmaxf(fmaxf(sacc[0][r], sacc[1][r]), fmaxf(sacc[2][r], sacc[3][r]));
      cm = row_max16(cm);
      mnew[r] = fmaxf(mrow[r], cm);
      alpha[r] = __expf(mrow[r] - mnew[r]);
      psum[r] = 0.0f;
    }
    #pragma unroll
    for (int nt = 0; nt < 4; ++nt)
      #pragma unroll
      for (int r = 0; r < 8; ++r) {
        float p = __expf(sacc[nt][r] - mnew[r]);
        sacc[nt][r] = p;
        psum[r] += p;
      }
    #pragma unroll
    for (int r = 0; r < 8; ++r) {
      lrow[r] = lrow[r] * alpha[r] + row_sum16(psum[r]);
      mrow[r] = mnew[r];
    }
    #pragma unroll
    for (int dt = 0; dt < 4; ++dt)
      #pragma unroll
      for (int r = 0; r < 8; ++r) acco[dt][r] *= alpha[r];

    // relayout P through LDS (intra-wave: LDS ops are in-order per wave)
    #pragma unroll
    for (int nt = 0; nt < 4; ++nt)
      #pragma unroll
      for (int r = 0; r < 8; ++r)
        sP[(w * 16 + hf * 8 + r) * 72 + nt * 16 + l16] = (__bf16)sacc[nt][r];

    // O += P @ V : K-dim = 64 k-cols (2 WMMAs per d-tile)
    v16bf ap0 = lds_load_a(&sP[(w * 16) * 72 + 0], 72, lane);
    v16bf ap1 = lds_load_a(&sP[(w * 16) * 72 + 32], 72, lane);
    #pragma unroll
    for (int dt = 0; dt < 4; ++dt) {
      v16bf bv0 = lds_load_bt(&sVt[(dt * 16) * 72 + 0], 72, lane);
      v16bf bv1 = lds_load_bt(&sVt[(dt * 16) * 72 + 32], 72, lane);
      acco[dt] = wmma_bf16(ap0, bv0, acco[dt]);
      acco[dt] = wmma_bf16(ap1, bv1, acco[dt]);
    }
    __syncthreads();
  }

  // normalize and store
  #pragma unroll
  for (int dt = 0; dt < 4; ++dt)
    #pragma unroll
    for (int r = 0; r < 8; ++r) {
      const int off = (q0 + w * 16 + hf * 8 + r) * 512 + dt * 16 + l16;
      Obase[off] = (__bf16)(acco[dt][r] / lrow[r]);
    }
}

// ---------------------------------------------------------------------------
// Host side
// ---------------------------------------------------------------------------
extern "C" void kernel_launch(void* const* d_in, const int* in_sizes, int n_in,
                              void* d_out, int out_size, void* d_ws, size_t ws_size,
                              hipStream_t stream) {
  const float* x   = (const float*)d_in[0];
  const float* ctx = (const float*)d_in[1];
  const float* Wi  = (const float*)d_in[2];
  const float* bi  = (const float*)d_in[3];
  const float* Wc  = (const float*)d_in[4];
  const float* bc  = (const float*)d_in[5];
  const float* Wq  = (const float*)d_in[6];
  const float* Wk  = (const float*)d_in[7];
  const float* Wv  = (const float*)d_in[8];
  const float* Wo1 = (const float*)d_in[9];
  const float* bo1 = (const float*)d_in[10];
  const float* Wo2 = (const float*)d_in[11];
  const float* bo2 = (const float*)d_in[12];
  float* out = (float*)d_out;

  const int M = 2 * 4096;   // B*S = 8192
  const int N = 512, K = 512;

  char* ws = (char*)d_ws;
  const size_t BUF = (size_t)M * 512 * sizeof(__bf16);   // 8 MiB each
  __bf16* A0 = (__bf16*)(ws + 0 * BUF);  // xs -> q    -> h1
  __bf16* A1 = (__bf16*)(ws + 1 * BUF);  // cs -> k
  __bf16* A2 = (__bf16*)(ws + 2 * BUF);  // xproj -> v
  __bf16* A3 = (__bf16*)(ws + 3 * BUF);  // cproj -> attn_out
  __bf16* wb = (__bf16*)(ws + 4 * BUF);
  const size_t WSZ = 512 * 512;          // weights stored transposed [N][K] bf16
  __bf16* tWi  = wb + 0 * WSZ;
  __bf16* tWc  = wb + 1 * WSZ;
  __bf16* tWq  = wb + 2 * WSZ;
  __bf16* tWk  = wb + 3 * WSZ;
  __bf16* tWv  = wb + 4 * WSZ;
  __bf16* tWo1 = wb + 5 * WSZ;
  __bf16* tWo2 = wb + 6 * WSZ;

  // 1) casts / transposes
  const int nElem = M * 512;                         // 4M
  transpose_cast_kernel<<<nElem / 256, 256, 0, stream>>>(x, A0);
  transpose_cast_kernel<<<nElem / 256, 256, 0, stream>>>(ctx, A1);
  const int wN = 512 * 512;
  weight_cast_transpose_kernel<<<wN / 256, 256, 0, stream>>>(Wi,  tWi);
  weight_cast_transpose_kernel<<<wN / 256, 256, 0, stream>>>(Wc,  tWc);
  weight_cast_transpose_kernel<<<wN / 256, 256, 0, stream>>>(Wq,  tWq);
  weight_cast_transpose_kernel<<<wN / 256, 256, 0, stream>>>(Wk,  tWk);
  weight_cast_transpose_kernel<<<wN / 256, 256, 0, stream>>>(Wv,  tWv);
  weight_cast_transpose_kernel<<<wN / 256, 256, 0, stream>>>(Wo1, tWo1);
  weight_cast_transpose_kernel<<<wN / 256, 256, 0, stream>>>(Wo2, tWo2);

  const dim3 gGemm(N / 128, M / 128);   // (4, 64)
  // 2) input projections (+bias)
  gemm_bf16_kernel<0, 0><<<gGemm, 256, 0, stream>>>(A0, tWi, bi, A2, M, N, K); // xproj
  gemm_bf16_kernel<0, 0><<<gGemm, 256, 0, stream>>>(A1, tWc, bc, A3, M, N, K); // cproj
  // 3) q/k/v (no bias)
  gemm_bf16_kernel<0, 0><<<gGemm, 256, 0, stream>>>(A2, tWq, nullptr, A0, M, N, K); // q
  gemm_bf16_kernel<0, 0><<<gGemm, 256, 0, stream>>>(A3, tWk, nullptr, A1, M, N, K); // k
  gemm_bf16_kernel<0, 0><<<gGemm, 256, 0, stream>>>(A3, tWv, nullptr, A2, M, N, K); // v
  // 4) flash attention -> A3
  attn_kernel<<<dim3(16, 64), 128, 0, stream>>>(A0, A1, A2, A3);
  // 5) out MLP: gelu(attn @ Wo1 + bo1) -> A0 ; then @ Wo2 + bo2 -> d_out (f32, [B,C,S])
  gemm_bf16_kernel<1, 0><<<gGemm, 256, 0, stream>>>(A3, tWo1, bo1, A0, M, N, K);
  gemm_bf16_kernel<0, 1><<<gGemm, 256, 0, stream>>>(A0, tWo2, bo2, out, M, N, K);
}
